// EdgeNeuralNet_20220706030181
// MI455X (gfx1250) — compile-verified, matrix operand order checked by reading, not run on hardware
//
#include <hip/hip_runtime.h>

// ---------------------------------------------------------------------------
// Problem dimensions (fixed by the reference)
// ---------------------------------------------------------------------------
#define B_DIM    8
#define S_DIM    512
#define ROWS     4096          // B*S
#define SPATIAL  512
#define HIDDEN   513
#define FBINS    257           // rfft bins
#define F2       514           // 2*FBINS
#define F4       1028          // 4*FBINS
#define IN_DIM   2052          // F2+F2+2*SPATIAL
#define GAMMA_H  128           // F//2
#define NSTEPS   20
#define TOLV     1e-3f

// K dims padded to multiples of 32 (zero-filled) so GEMM needs no K checks
#define KP_IN    2080          // IN_DIM -> 65 chunks
#define KP_F4    1056          // F4     -> 33 chunks
#define KP_F2    544           // F2     -> 17 chunks
// Weight^T row counts padded to multiples of BN=128 (zero rows)
#define NB_F4    1152
#define NB_F2    640
#define NB_FB    384
#define NB_GH    128

typedef _Float16 half_t;
typedef __attribute__((ext_vector_type(16))) _Float16 v16h;
typedef __attribute__((ext_vector_type(8)))  _Float16 v8h;
typedef __attribute__((ext_vector_type(8)))  float    v8f;
typedef __attribute__((ext_vector_type(4)))  int      v4i;

#define INV_SQRT2 0.70710678118654752f
#define TWO_PI    6.28318530717958648f

// CDNA5 async global->LDS staging (ASYNCcnt-tracked), if the toolchain exposes it
#if defined(__has_builtin)
#if __has_builtin(__builtin_amdgcn_global_load_async_to_lds_b128) && \
    __has_builtin(__builtin_amdgcn_s_wait_asynccnt)
#define HAVE_ASYNC_LDS 1
#endif
#endif
#ifndef HAVE_ASYNC_LDS
#define HAVE_ASYNC_LDS 0
#endif

#if HAVE_ASYNC_LDS
// builtin signature (from compiler diagnostic): (v4i __device__*, v4i __shared__*, imm, imm)
typedef __attribute__((address_space(1))) v4i* as1_v4i;
typedef __attribute__((address_space(3))) v4i* as3_v4i;
__device__ inline void async_b128(const half_t* g, half_t* l) {
    __builtin_amdgcn_global_load_async_to_lds_b128(
        (as1_v4i)(void*)(const_cast<half_t*>(g)), (as3_v4i)(void*)l, 0, 0);
}
#endif

// ---------------------------------------------------------------------------
// WMMA f16 GEMM:  C[M,N] = A[M,K] * W[K,N] + bias
//   A  : f16 row-major,       pitch lda  (mult of 8), K zero-padded to K32
//   WT : f16 W^T [Npad][K32], pitch ldwt (mult of 8), zero pad rows/cols
// Block: 256 threads = 8 waves; tile 128x128, K chunk 32.
// Wave grid 4(M) x 2(N): each wave -> 32x64 = 2x4 WMMA tiles, f32 accum.
// Double-buffered LDS; staging is async global->LDS when available, else
// register-staged global_load_b128 -> ds_store_b128.
// ---------------------------------------------------------------------------
#define BM 128
#define BN 128
#define BK 32
#define LDST 40                 // LDS row stride in halfs (80B: 16B-aligned, bank-spread)

__device__ inline v16h load_frag(const half_t* rowbase, int khalf) {
    // elements 0..7  -> K = khalf + e        (VGPR0..3)
    // elements 8..15 -> K = 16 + khalf + e-8 (VGPR4..7)
    v8h lo = *(const v8h*)(rowbase + khalf);
    v8h hi = *(const v8h*)(rowbase + 16 + khalf);
    v16h f;
#pragma unroll
    for (int e = 0; e < 8; ++e) { f[e] = lo[e]; f[8 + e] = hi[e]; }
    return f;
}

__global__ __launch_bounds__(256) void k_gemm(
    const half_t* __restrict__ A, const half_t* __restrict__ WT,
    const float* __restrict__ bias, float* __restrict__ C,
    int K32, int lda, int ldwt, int N, int ldc)
{
    __shared__ __align__(16) half_t sA[2][BM][LDST];
    __shared__ __align__(16) half_t sB[2][BN][LDST];

    const int tid   = threadIdx.x;
    const int lane  = tid & 31;
    const int wave  = tid >> 5;
    const int wm    = wave & 3;          // row offset wm*32
    const int wn    = wave >> 2;         // col offset wn*64
    const int l16   = lane & 15;
    const int khalf = (lane >> 4) * 8;   // 0 or 8
    const int bm0   = blockIdx.x * BM;
    const int bn0   = blockIdx.y * BN;

    // staging: tiles are 128 rows x 32 cols = 512 v8h vectors; 2 per thread
    const int r0 = tid >> 2;             // 0..63
    const int c0 = (tid & 3) * 8;        // 0,8,16,24
    const half_t* Ar0 = A  + (size_t)(bm0 + r0)      * lda  + c0;
    const half_t* Ar1 = A  + (size_t)(bm0 + r0 + 64) * lda  + c0;
    const half_t* Br0 = WT + (size_t)(bn0 + r0)      * ldwt + c0;
    const half_t* Br1 = WT + (size_t)(bn0 + r0 + 64) * ldwt + c0;

    v8f acc[2][4];
#pragma unroll
    for (int i = 0; i < 2; ++i)
#pragma unroll
        for (int j = 0; j < 4; ++j) { v8f z = {}; acc[i][j] = z; }

    const int nch = K32 >> 5;

#if HAVE_ASYNC_LDS
    // ---- async global->LDS double-buffered pipeline ----
    {
        // prologue: chunk 0 into buffer 0
        async_b128(Ar0, &sA[0][r0][c0]);
        async_b128(Ar1, &sA[0][r0 + 64][c0]);
        async_b128(Br0, &sB[0][r0][c0]);
        async_b128(Br1, &sB[0][r0 + 64][c0]);
        __builtin_amdgcn_s_wait_asynccnt(0);
        __syncthreads();

        for (int c = 0; c < nch; ++c) {
            const int p = c & 1;
            const bool more = (c + 1 < nch);
            if (more) {                   // stage next chunk into the other buffer
                const int ko = (c + 1) << 5;
                const int q = p ^ 1;
                async_b128(Ar0 + ko, &sA[q][r0][c0]);
                async_b128(Ar1 + ko, &sA[q][r0 + 64][c0]);
                async_b128(Br0 + ko, &sB[q][r0][c0]);
                async_b128(Br1 + ko, &sB[q][r0 + 64][c0]);
            }

            v16h afrag[2], bfrag[4];
#pragma unroll
            for (int i = 0; i < 2; ++i)
                afrag[i] = load_frag(&sA[p][wm * 32 + i * 16 + l16][0], khalf);
#pragma unroll
            for (int j = 0; j < 4; ++j)
                bfrag[j] = load_frag(&sB[p][wn * 64 + j * 16 + l16][0], khalf);

#pragma unroll
            for (int i = 0; i < 2; ++i)
#pragma unroll
                for (int j = 0; j < 4; ++j)
                    acc[i][j] = __builtin_amdgcn_wmma_f32_16x16x32_f16(
                        false, afrag[i], false, bfrag[j],
                        (short)0, acc[i][j], false, false);

            if (more) {
                __builtin_amdgcn_s_wait_asynccnt(0);
                __syncthreads();
            }
        }
    }
#else
    // ---- register-staged double-buffered pipeline ----
    {
        v8h ra0 = *(const v8h*)Ar0;
        v8h ra1 = *(const v8h*)Ar1;
        v8h rb0 = *(const v8h*)Br0;
        v8h rb1 = *(const v8h*)Br1;
        *(v8h*)&sA[0][r0][c0]      = ra0;
        *(v8h*)&sA[0][r0 + 64][c0] = ra1;
        *(v8h*)&sB[0][r0][c0]      = rb0;
        *(v8h*)&sB[0][r0 + 64][c0] = rb1;
        __syncthreads();

        for (int c = 0; c < nch; ++c) {
            const int p = c & 1;
            const bool more = (c + 1 < nch);
            if (more) {
                const int ko = (c + 1) << 5;
                ra0 = *(const v8h*)(Ar0 + ko);
                ra1 = *(const v8h*)(Ar1 + ko);
                rb0 = *(const v8h*)(Br0 + ko);
                rb1 = *(const v8h*)(Br1 + ko);
            }

            v16h afrag[2], bfrag[4];
#pragma unroll
            for (int i = 0; i < 2; ++i)
                afrag[i] = load_frag(&sA[p][wm * 32 + i * 16 + l16][0], khalf);
#pragma unroll
            for (int j = 0; j < 4; ++j)
                bfrag[j] = load_frag(&sB[p][wn * 64 + j * 16 + l16][0], khalf);

#pragma unroll
            for (int i = 0; i < 2; ++i)
#pragma unroll
                for (int j = 0; j < 4; ++j)
                    acc[i][j] = __builtin_amdgcn_wmma_f32_16x16x32_f16(
                        false, afrag[i], false, bfrag[j],
                        (short)0, acc[i][j], false, false);

            if (more) {
                const int q = p ^ 1;
                __syncthreads();
                *(v8h*)&sA[q][r0][c0]      = ra0;
                *(v8h*)&sA[q][r0 + 64][c0] = ra1;
                *(v8h*)&sB[q][r0][c0]      = rb0;
                *(v8h*)&sB[q][r0 + 64][c0] = rb1;
                __syncthreads();
            }
        }
    }
#endif

    // epilogue: row = tile_m + r + khalf, col = tile_n + l16
#pragma unroll
    for (int i = 0; i < 2; ++i) {
        const int rbase = bm0 + wm * 32 + i * 16 + khalf;
#pragma unroll
        for (int j = 0; j < 4; ++j) {
            const int col = bn0 + wn * 64 + j * 16 + l16;
            if (col < N) {
                const float bv = bias[col];
#pragma unroll
                for (int r = 0; r < 8; ++r)
                    C[(size_t)(rbase + r) * ldc + col] = acc[i][j][r] + bv;
            }
        }
    }
}

// ---------------------------------------------------------------------------
// rFFT of 512-point real rows -> interleaved (re,im) [rows, 514]
// ---------------------------------------------------------------------------
__global__ __launch_bounds__(256) void k_rfft(const float* __restrict__ x,
                                              float* __restrict__ out)
{
    __shared__ float sx[SPATIAL];
    int row = blockIdx.x;
    const float* xr = x + (size_t)row * SPATIAL;
    for (int i = threadIdx.x; i < SPATIAL; i += 256) sx[i] = xr[i];
    __syncthreads();
    for (int k = threadIdx.x; k < FBINS; k += 256) {
        float re = 0.f, im = 0.f;
        for (int n = 0; n < SPATIAL; ++n) {
            int m = (k * n) & (SPATIAL - 1);          // exact angle reduction
            float ang = -TWO_PI * (float)m * (1.f / SPATIAL);
            float sn, cs;
            sincosf(ang, &sn, &cs);
            re += sx[n] * cs;
            im += sx[n] * sn;
        }
        out[(size_t)row * F2 + 2 * k]     = re;
        out[(size_t)row * F2 + 2 * k + 1] = im;
    }
}

// irFFT: x[n] = (1/512)(re0 + 2*sum_{k=1..255}(re*cos - im*sin) + (-1)^n re256)
__global__ __launch_bounds__(256) void k_irfft(const float* __restrict__ cf,
                                               float* __restrict__ out)
{
    __shared__ float re[FBINS], im[FBINS];
    int row = blockIdx.x;
    const float* p = cf + (size_t)row * F2;
    for (int i = threadIdx.x; i < FBINS; i += 256) { re[i] = p[2 * i]; im[i] = p[2 * i + 1]; }
    __syncthreads();
    for (int n = threadIdx.x; n < SPATIAL; n += 256) {
        float s = re[0] + ((n & 1) ? -re[FBINS - 1] : re[FBINS - 1]);
        for (int k = 1; k < FBINS - 1; ++k) {
            int m = (k * n) & (SPATIAL - 1);
            float ang = TWO_PI * (float)m * (1.f / SPATIAL);
            float sn, cs;
            sincosf(ang, &sn, &cs);
            s += 2.f * (re[k] * cs - im[k] * sn);
        }
        out[(size_t)row * SPATIAL + n] = s * (1.f / SPATIAL);
    }
}

// ---------------------------------------------------------------------------
// Packing / conversion kernels (write zero into all pad regions)
// ---------------------------------------------------------------------------
// W[K,N] f32 -> WT[NB][K32] f16 transposed, zero-padded
__global__ void k_cvt_wT(const float* __restrict__ W, half_t* __restrict__ WT,
                         int K, int N, int K32, int NB) {
    int idx = blockIdx.x * blockDim.x + threadIdx.x;
    if (idx >= NB * K32) return;
    int n = idx / K32;
    int k = idx - n * K32;
    float v = (n < N && k < K) ? W[(size_t)k * N + n] : 0.f;
    WT[idx] = (half_t)v;
}

__global__ void k_pack_combined(const float* __restrict__ c, const float* __restrict__ t,
                                const float* __restrict__ sp, const float* __restrict__ tp,
                                half_t* __restrict__ comb) {
    int idx = blockIdx.x * blockDim.x + threadIdx.x;
    if (idx >= ROWS * KP_IN) return;
    int row = idx / KP_IN;
    int col = idx - row * KP_IN;
    int b = row >> 9;
    float v = 0.f;
    if (col < F2)                          v = c[(size_t)row * F2 + col];
    else if (col < 2 * F2)                 v = t[(size_t)row * F2 + (col - F2)];
    else if (col < 2 * F2 + SPATIAL)       v = sp[(size_t)b * HIDDEN + (col - 2 * F2)];
    else if (col < IN_DIM)                 v = tp[(size_t)b * HIDDEN + (col - 2 * F2 - SPATIAL)];
    comb[idx] = (half_t)v;
}

__global__ void k_pack_safe(const float* __restrict__ x, half_t* __restrict__ o) {
    int idx = blockIdx.x * blockDim.x + threadIdx.x;
    if (idx >= ROWS * KP_F2) return;
    int row = idx / KP_F2;
    int col = idx - row * KP_F2;
    float v = 0.f;
    if (col < F2) {
        v = x[(size_t)row * F2 + col];
        if (isnan(v)) v = 0.f;
        else if (v >  1e6f) v =  1e6f;
        else if (v < -1e6f) v = -1e6f;
        v = fminf(fmaxf(v, -60000.f), 60000.f);   // keep finite in f16
    }
    o[idx] = (half_t)v;
}

__global__ void k_tanh_inplace(float* __restrict__ x, int n) {
    int i = blockIdx.x * blockDim.x + threadIdx.x;
    if (i < n) x[i] = tanhf(x[i]);
}

// LayerNorm + exact GELU, f16 out with padded pitch; one block per row
__global__ __launch_bounds__(256) void k_ln_gelu(const float* __restrict__ pre,
                                                 const float* __restrict__ g,
                                                 const float* __restrict__ beta,
                                                 half_t* __restrict__ h, int N, int HP) {
    __shared__ float sh[256];
    int row = blockIdx.x, tid = threadIdx.x;
    const float* p = pre + (size_t)row * N;
    float s = 0.f;
    for (int i = tid; i < N; i += 256) s += p[i];
    sh[tid] = s; __syncthreads();
    for (int k = 128; k > 0; k >>= 1) { if (tid < k) sh[tid] += sh[tid + k]; __syncthreads(); }
    float mean = sh[0] / (float)N; __syncthreads();
    float s2 = 0.f;
    for (int i = tid; i < N; i += 256) { float d = p[i] - mean; s2 += d * d; }
    sh[tid] = s2; __syncthreads();
    for (int k = 128; k > 0; k >>= 1) { if (tid < k) sh[tid] += sh[tid + k]; __syncthreads(); }
    float inv = rsqrtf(sh[0] / (float)N + 1e-5f);
    for (int i = tid; i < HP; i += 256) {
        float ge = 0.f;
        if (i < N) {
            float y = (p[i] - mean) * inv * g[i] + beta[i];
            ge = 0.5f * y * (1.f + erff(y * INV_SQRT2));
        }
        h[(size_t)row * HP + i] = (half_t)ge;
    }
}

// out[row] = sigmoid( sum_j gelu(hid[row,j]) * w2[j] + b2 ) [+ addv]
__global__ __launch_bounds__(256) void k_gelu_dot(const float* __restrict__ hid,
                                                  const float* __restrict__ w2,
                                                  const float* __restrict__ b2,
                                                  float* __restrict__ outv,
                                                  int H, float addv, int nanfix) {
    __shared__ float sh[256];
    int row = blockIdx.x, tid = threadIdx.x;
    const float* p = hid + (size_t)row * H;
    float s = 0.f;
    for (int i = tid; i < H; i += 256) {
        float y = p[i];
        float ge = 0.5f * y * (1.f + erff(y * INV_SQRT2));
        s += ge * w2[i];
    }
    sh[tid] = s; __syncthreads();
    for (int k = 128; k > 0; k >>= 1) { if (tid < k) sh[tid] += sh[tid + k]; __syncthreads(); }
    if (tid == 0) {
        float z = sh[0] + b2[0];
        float sg = 1.f / (1.f + expf(-z));
        if (nanfix && isnan(sg)) sg = 0.f;
        outv[row] = sg + addv;
    }
}

// state update + reductions; one block per row
__global__ __launch_bounds__(256) void k_update(
    const float* __restrict__ c,  const float* __restrict__ t,
    const float* __restrict__ fc, const float* __restrict__ ft,
    const float* __restrict__ ac, const float* __restrict__ at,
    const float* __restrict__ gm, const float* __restrict__ fd,
    const float* __restrict__ cs, const float* __restrict__ ts,
    float* __restrict__ nc, float* __restrict__ nt, float* __restrict__ acc) {
    __shared__ float sh[256];
    int row = blockIdx.x, tid = threadIdx.x;
    float gc = gm[row] * ac[row] * cs[0];
    float gt = gm[row] * at[row] * ts[0];
    size_t base = (size_t)row * F2;
    float d2c = 0.f, d2t = 0.f, s2c = 0.f, s2t = 0.f;
    for (int j = tid; j < F2; j += 256) {
        float damp = fd[(j < FBINS) ? j : (j - FBINS)];  // tiled, per spec
        float oc = c[base + j];
        float vc = damp * oc + gc * fc[base + j];
        nc[base + j] = vc;
        float dd = vc - oc; d2c += dd * dd; s2c += vc * vc;
        float ot = t[base + j];
        float vt = damp * ot + gt * ft[base + j];
        nt[base + j] = vt;
        dd = vt - ot; d2t += dd * dd; s2t += vt * vt;
    }
    float vals[4] = { d2c, d2t, s2c, s2t };
#pragma unroll
    for (int q = 0; q < 4; ++q) {
        sh[tid] = vals[q]; __syncthreads();
        for (int k = 128; k > 0; k >>= 1) { if (tid < k) sh[tid] += sh[tid + k]; __syncthreads(); }
        if (tid == 0) atomicAdd(&acc[q], (q < 2) ? sqrtf(sh[0]) : sh[0]);
        __syncthreads();
    }
}

// flags = [done, commit, fac_c, fac_t]
__global__ void k_finalize(float* __restrict__ acc, float* __restrict__ flags) {
    float dc = acc[0] * (1.f / ROWS);
    float dt = acc[1] * (1.f / ROWS);
    int done = (flags[0] != 0.f);
    int conv = (dc < TOLV) && (dt < TOLV);
    if (done) {
        flags[1] = 0.f;
    } else {
        flags[1] = 1.f;
        float nrc = sqrtf(acc[2]);
        float nrt = sqrtf(acc[3]);
        flags[2] = conv ? 1.f : ((nrc > 10.f) ? 10.f / nrc : 1.f);
        flags[3] = conv ? 1.f : ((nrt > 10.f) ? 10.f / nrt : 1.f);
        if (conv) flags[0] = 1.f;
    }
}

__global__ void k_apply(float* __restrict__ c, const float* __restrict__ nc,
                        float* __restrict__ t, const float* __restrict__ nt,
                        const float* __restrict__ flags, int n) {
    int i = blockIdx.x * blockDim.x + threadIdx.x;
    if (i >= n) return;
    if (flags[1] == 0.f) return;  // uniform branch
    c[i] = nc[i] * flags[2];
    t[i] = nt[i] * flags[3];
}

__global__ void k_zero_acc(float* __restrict__ acc) {
    if (threadIdx.x < 4) acc[threadIdx.x] = 0.f;
}
__global__ void k_init_flags(float* __restrict__ flags) {
    if (threadIdx.x < 4) flags[threadIdx.x] = 0.f;
}

// ---------------------------------------------------------------------------
// Host-side orchestration
// ---------------------------------------------------------------------------
extern "C" void kernel_launch(void* const* d_in, const int* in_sizes, int n_in,
                              void* d_out, int out_size, void* d_ws, size_t ws_size,
                              hipStream_t stream) {
    (void)in_sizes; (void)n_in; (void)out_size; (void)ws_size;

    const float* carrier_in = (const float*)d_in[0];
    const float* traj_in    = (const float*)d_in[1];
    const float* src_pos    = (const float*)d_in[2];
    const float* tgt_pos    = (const float*)d_in[3];
    const float* cw1 = (const float*)d_in[4];
    const float* cb1 = (const float*)d_in[5];
    const float* cg  = (const float*)d_in[6];
    const float* cbe = (const float*)d_in[7];
    const float* cw2 = (const float*)d_in[8];
    const float* cb2 = (const float*)d_in[9];
    const float* tw1 = (const float*)d_in[10];
    const float* tb1 = (const float*)d_in[11];
    const float* tg  = (const float*)d_in[12];
    const float* tbe = (const float*)d_in[13];
    const float* tw2 = (const float*)d_in[14];
    const float* tb2 = (const float*)d_in[15];
    const float* fd  = (const float*)d_in[16];
    const float* dw1 = (const float*)d_in[17];
    const float* db1 = (const float*)d_in[18];
    const float* dw2 = (const float*)d_in[19];
    const float* db2 = (const float*)d_in[20];
    const float* sw1 = (const float*)d_in[21];
    const float* sb1 = (const float*)d_in[22];
    const float* sw2 = (const float*)d_in[23];
    const float* sb2 = (const float*)d_in[24];
    const float* cscale = (const float*)d_in[25];
    const float* tscale = (const float*)d_in[26];

    char* base = (char*)d_ws;
    size_t off = 0;
    auto alloc = [&](size_t bytes) -> char* {
        char* p = base + off;
        off = (off + bytes + 255) & ~(size_t)255;
        return p;
    };
    float*  c_st   = (float*)alloc((size_t)ROWS * F2 * 4);
    float*  t_st   = (float*)alloc((size_t)ROWS * F2 * 4);
    float*  nc_st  = (float*)alloc((size_t)ROWS * F2 * 4);
    float*  nt_st  = (float*)alloc((size_t)ROWS * F2 * 4);
    float*  fc_b   = (float*)alloc((size_t)ROWS * F2 * 4);
    float*  ft_b   = (float*)alloc((size_t)ROWS * F2 * 4);
    float*  pre1   = (float*)alloc((size_t)ROWS * F4 * 4);
    half_t* h16    = (half_t*)alloc((size_t)ROWS * KP_F4 * 2);
    half_t* comb16 = (half_t*)alloc((size_t)ROWS * KP_IN * 2);
    half_t* safc16 = (half_t*)alloc((size_t)ROWS * KP_F2 * 2);
    half_t* saft16 = (half_t*)alloc((size_t)ROWS * KP_F2 * 2);
    float*  hid    = (float*)alloc((size_t)ROWS * FBINS * 4);
    float*  alpc   = (float*)alloc(ROWS * 4);
    float*  alpt   = (float*)alloc(ROWS * 4);
    float*  gamv   = (float*)alloc(ROWS * 4);
    half_t* cw1h   = (half_t*)alloc((size_t)NB_F4 * KP_IN * 2);
    half_t* tw1h   = (half_t*)alloc((size_t)NB_F4 * KP_IN * 2);
    half_t* cw2h   = (half_t*)alloc((size_t)NB_F2 * KP_F4 * 2);
    half_t* tw2h   = (half_t*)alloc((size_t)NB_F2 * KP_F4 * 2);
    half_t* dw1h   = (half_t*)alloc((size_t)NB_FB * KP_F2 * 2);
    half_t* sw1h   = (half_t*)alloc((size_t)NB_GH * KP_F2 * 2);
    float*  acc    = (float*)alloc(64);
    float*  flags  = (float*)alloc(64);

    auto cvtT = [&](const float* W, half_t* WT, int K, int N, int K32, int NB) {
        int total = NB * K32;
        k_cvt_wT<<<(total + 255) / 256, 256, 0, stream>>>(W, WT, K, N, K32, NB);
    };
    cvtT(cw1, cw1h, IN_DIM, F4, KP_IN, NB_F4);
    cvtT(tw1, tw1h, IN_DIM, F4, KP_IN, NB_F4);
    cvtT(cw2, cw2h, F4, F2, KP_F4, NB_F2);
    cvtT(tw2, tw2h, F4, F2, KP_F4, NB_F2);
    cvtT(dw1, dw1h, F2, FBINS, KP_F2, NB_FB);
    cvtT(sw1, sw1h, F2, GAMMA_H, KP_F2, NB_GH);

    k_rfft<<<ROWS, 256, 0, stream>>>(carrier_in, c_st);
    k_rfft<<<ROWS, 256, 0, stream>>>(traj_in, t_st);
    k_init_flags<<<1, 32, 0, stream>>>(flags);

    auto gemm = [&](const half_t* A, int lda, const half_t* WT, int ldwt,
                    const float* bias, float* C, int ldc, int K32, int N, int NB) {
        dim3 g(ROWS / BM, NB / BN);
        k_gemm<<<g, 256, 0, stream>>>(A, WT, bias, C, K32, lda, ldwt, N, ldc);
    };

    const int nF2  = ROWS * F2;
    const int gF2  = (nF2 + 255) / 256;
    const int gSAF = (ROWS * KP_F2 + 255) / 256;
    const int gCMB = (ROWS * KP_IN + 255) / 256;

    for (int step = 0; step < NSTEPS; ++step) {
        k_zero_acc<<<1, 32, 0, stream>>>(acc);
        k_pack_combined<<<gCMB, 256, 0, stream>>>(c_st, t_st, src_pos, tgt_pos, comb16);

        // carrier freq net
        gemm(comb16, KP_IN, cw1h, KP_IN, cb1, pre1, F4, KP_IN, F4, NB_F4);
        k_ln_gelu<<<ROWS, 256, 0, stream>>>(pre1, cg, cbe, h16, F4, KP_F4);
        gemm(h16, KP_F4, cw2h, KP_F4, cb2, fc_b, F2, KP_F4, F2, NB_F2);
        k_tanh_inplace<<<gF2, 256, 0, stream>>>(fc_b, nF2);

        // trajectory freq net
        gemm(comb16, KP_IN, tw1h, KP_IN, tb1, pre1, F4, KP_IN, F4, NB_F4);
        k_ln_gelu<<<ROWS, 256, 0, stream>>>(pre1, tg, tbe, h16, F4, KP_F4);
        gemm(h16, KP_F4, tw2h, KP_F4, tb2, ft_b, F2, KP_F4, F2, NB_F2);
        k_tanh_inplace<<<gF2, 256, 0, stream>>>(ft_b, nF2);

        // damping / scale MLPs on nan-sanitized state
        k_pack_safe<<<gSAF, 256, 0, stream>>>(c_st, safc16);
        k_pack_safe<<<gSAF, 256, 0, stream>>>(t_st, saft16);

        gemm(safc16, KP_F2, dw1h, KP_F2, db1, hid, FBINS, KP_F2, FBINS, NB_FB);
        k_gelu_dot<<<ROWS, 256, 0, stream>>>(hid, dw2, db2, alpc, FBINS, 0.f, 1);
        gemm(saft16, KP_F2, dw1h, KP_F2, db1, hid, FBINS, KP_F2, FBINS, NB_FB);
        k_gelu_dot<<<ROWS, 256, 0, stream>>>(hid, dw2, db2, alpt, FBINS, 0.f, 1);
        gemm(safc16, KP_F2, sw1h, KP_F2, sb1, hid, GAMMA_H, KP_F2, GAMMA_H, NB_GH);
        k_gelu_dot<<<ROWS, 256, 0, stream>>>(hid, sw2, sb2, gamv, GAMMA_H, 0.5f, 0);

        // state update + convergence / clamp logic
        k_update<<<ROWS, 256, 0, stream>>>(c_st, t_st, fc_b, ft_b, alpc, alpt, gamv,
                                           fd, cscale, tscale, nc_st, nt_st, acc);
        k_finalize<<<1, 1, 0, stream>>>(acc, flags);
        k_apply<<<gF2, 256, 0, stream>>>(c_st, nc_st, t_st, nt_st, flags, nF2);
    }

    float* out = (float*)d_out;
    k_irfft<<<ROWS, 256, 0, stream>>>(c_st, out);
    k_irfft<<<ROWS, 256, 0, stream>>>(t_st, out + (size_t)ROWS * SPATIAL);
}
